// TransformerBlock_39144331935812
// MI455X (gfx1250) — matrix operand order repaired and checked by reading, not compiled, not run on hardware
//
#include <hip/hip_runtime.h>
#include <cstdint>

typedef __bf16 bf16_t;
typedef __attribute__((ext_vector_type(16))) __bf16 v16bf;
typedef __attribute__((ext_vector_type(8)))  float  v8f;

union U32H { uint32_t u; bf16_t h[2]; };

static __device__ __forceinline__ v8f wmma_bf16(v16bf a, v16bf b, v8f c) {
  return __builtin_amdgcn_wmma_f32_16x16x32_bf16(false, a, false, b, (short)0, c, false, false);
}
// same A operand as the immediately preceding identical WMMA -> reuse-A hint (ISA 7.12 OPSEL.RA)
static __device__ __forceinline__ v8f wmma_bf16_ra(v16bf a, v16bf b, v8f c) {
  return __builtin_amdgcn_wmma_f32_16x16x32_bf16(false, a, false, b, (short)0, c, true, false);
}

// A-fragment (ISA 7.12.2): lane = m + 16*halfA, element e; K = 16*(e>>3)+8*halfA+(e&7)
//   inverse: halfA = (k>>3)&1, e = 8*(k>>4) + (k&7)
// B-fragment: lane = n + 16*halfB, element e; K = 16*(e>>3)+8*halfB+(e&7) (same order as A)
// D-fragment: vgpr r, lane = col + 16*half, row = r + 8*half

// ----------------------- elementwise helpers (vectorized) -----------------------
__global__ void __launch_bounds__(256) k_copy_f32v(const float4* __restrict__ s,
                                                   float4* __restrict__ d, int n4) {
  int i = blockIdx.x * 256 + threadIdx.x;
  if (i < n4) d[i] = s[i];
}

__global__ void __launch_bounds__(256) k_f32_to_bf16v(const float4* __restrict__ s,
                                                      bf16_t* __restrict__ d, int n4) {
  int i = blockIdx.x * 256 + threadIdx.x;
  if (i < n4) {
    float4 v = s[i];
    bf16_t t[4] = {(bf16_t)v.x, (bf16_t)v.y, (bf16_t)v.z, (bf16_t)v.w};
    *(uint2*)(d + 4 * (long long)i) = *(uint2*)t;
  }
}

// ctx: [4][77][768] (token-major) -> ctxT: [4][768][128] bf16, cols 77..127 zero-padded
__global__ void __launch_bounds__(256) k_ctx_transpose(const float* __restrict__ ctx,
                                                       bf16_t* __restrict__ out) {
  int i = blockIdx.x * 256 + threadIdx.x;
  if (i >= 4 * 768 * 128) return;
  int b = i / (768 * 128);
  int r = i % (768 * 128);
  int c = r >> 7;
  int s = r & 127;
  float v = (s < 77) ? ctx[(long long)b * 77 * 768 + (long long)s * 768 + c] : 0.0f;
  out[i] = (bf16_t)v;
}

// ----------------------- GroupNorm (B=4, C=512, G=8, N=1024) -----------------------
__global__ void __launch_bounds__(256) k_groupnorm(const float* __restrict__ x,
                                                   const float* __restrict__ gw,
                                                   const float* __restrict__ gb,
                                                   bf16_t* __restrict__ out) {
  __shared__ float red[256];
  __shared__ float s_mean, s_inv;
  const int b = blockIdx.x >> 3;
  const int g = blockIdx.x & 7;
  const long long base = (long long)b * 512 * 1024 + (long long)g * 64 * 1024;
  const float4* x4 = (const float4*)(x + base);
  float sum = 0.f, ss = 0.f;
  for (int i = threadIdx.x; i < 16384; i += 256) {
    float4 v = x4[i];
    sum += v.x + v.y + v.z + v.w;
    ss += v.x * v.x + v.y * v.y + v.z * v.z + v.w * v.w;
  }
  red[threadIdx.x] = sum;
  __syncthreads();
  for (int s = 128; s > 0; s >>= 1) {
    if (threadIdx.x < s) red[threadIdx.x] += red[threadIdx.x + s];
    __syncthreads();
  }
  float tot = red[0];
  __syncthreads();
  red[threadIdx.x] = ss;
  __syncthreads();
  for (int s = 128; s > 0; s >>= 1) {
    if (threadIdx.x < s) red[threadIdx.x] += red[threadIdx.x + s];
    __syncthreads();
  }
  float tot2 = red[0];
  if (threadIdx.x == 0) {
    float mean = tot * (1.0f / 65536.0f);
    float var = tot2 * (1.0f / 65536.0f) - mean * mean;
    s_mean = mean;
    s_inv = rsqrtf(var + 1e-5f);
  }
  __syncthreads();
  float mean = s_mean, inv = s_inv;
  for (int i = threadIdx.x; i < 16384; i += 256) {
    int c = (g << 6) + (i >> 8);
    float w = gw[c] * inv, bb = gb[c] - mean * w;
    float4 v = x4[i];
    bf16_t t[4] = {(bf16_t)(v.x * w + bb), (bf16_t)(v.y * w + bb), (bf16_t)(v.z * w + bb),
                   (bf16_t)(v.w * w + bb)};
    *(uint2*)(out + base + 4 * (long long)i) = *(uint2*)t;
  }
}

// ----------------------- bf16 WMMA GEMM (128x64 tile, double-buffered frag LDS) -------------
// Out[b,m,n] = sum_k A[m,k]*Bm[b,k,n] (+bias)(+GELU)(+res). Requires M%128==0, N%64==0, K%32==0.
// 8 waves; wave w owns m-rows [w*16, w*16+16) x all 64 n: 4 WMMA per k-chunk sharing one A frag.
// Software pipeline: unguarded global loads for chunk c+1 after the barrier, WMMA on c,
// LDS stores of c+1 after compute. One barrier per chunk.
__global__ void __launch_bounds__(256) k_gemm(const bf16_t* __restrict__ A,
                                              const bf16_t* __restrict__ Bm,
                                              const float* __restrict__ bias,
                                              const float* res, float* outF, bf16_t* outB,
                                              int M, int N, int K, int act) {
  __shared__ v16bf AfV[2][8][32];  // [buf][m-subtile][lane]
  __shared__ v16bf BfV[2][4][32];  // [buf][n-subtile][lane]
  const int tid = threadIdx.x;
  const int lane = tid & 31;
  const int wave = tid >> 5;
  const int half = lane >> 4;
  const int l16 = lane & 15;
  const int n0 = blockIdx.x * 64;
  const int m0 = blockIdx.y * 128;
  const int b = blockIdx.z;
  const bf16_t* Bb = Bm + (long long)b * K * N;
  v8f acc[4] = {};
  uint32_t va[8], vb[4];

  auto gload = [&](int k0) {
#pragma unroll
    for (int j = 0; j < 8; ++j) {  // A: 128x32 as 4B pairs
      int i = tid + j * 256;
      int m = i >> 4, kk = (i & 15) * 2;
      va[j] = *(const uint32_t*)(A + (long long)(m0 + m) * K + k0 + kk);
    }
#pragma unroll
    for (int j = 0; j < 4; ++j) {  // B: 32x64 as 4B pairs (N%64==0: no guards)
      int i = tid + j * 256;
      int krel = i >> 5, np = (i & 31) * 2;
      vb[j] = *(const uint32_t*)(Bb + (long long)(k0 + krel) * N + n0 + np);
    }
  };
  auto lstore = [&](int buf) {
    bf16_t* Afp = (bf16_t*)AfV[buf];
    bf16_t* Bfp = (bf16_t*)BfV[buf];
#pragma unroll
    for (int j = 0; j < 8; ++j) {
      int i = tid + j * 256;
      int m = i >> 4, kk = (i & 15) * 2;
      int laneA = (m & 15) + 16 * ((kk >> 3) & 1);
      int e = 8 * (kk >> 4) + (kk & 7);
      *(uint32_t*)(Afp + (((m >> 4) * 32 + laneA) * 16 + e)) = va[j];
    }
#pragma unroll
    for (int j = 0; j < 4; ++j) {
      int i = tid + j * 256;
      int krel = i >> 5, np = (i & 31) * 2;
      int halfB = (krel >> 3) & 1, e = 8 * (krel >> 4) + (krel & 7);
      U32H u;
      u.u = vb[j];
      Bfp[(((np >> 4) * 32) + (np & 15) + 16 * halfB) * 16 + e] = u.h[0];
      int np1 = np + 1;
      Bfp[(((np1 >> 4) * 32) + (np1 & 15) + 16 * halfB) * 16 + e] = u.h[1];
    }
  };

  const int NC = K >> 5;
  gload(0);
  lstore(0);
  for (int c = 0; c < NC; ++c) {
    __syncthreads();  // buf (c&1) visible; buf (c+1)&1 free (its readers finished last iter)
    if (c + 1 < NC) {
      gload((c + 1) * 32);  // HBM latency hides under WMMA below
      if (c + 2 < NC)
        __builtin_prefetch(A + (long long)(m0 + (tid >> 1)) * K + (c + 2) * 32, 0, 3);
    }
    int buf = c & 1;
    v16bf af = AfV[buf][wave][lane];  // 32B contiguous ds_load per lane
    v16bf b0 = BfV[buf][0][lane];
    v16bf b1 = BfV[buf][1][lane];
    v16bf b2 = BfV[buf][2][lane];
    v16bf b3 = BfV[buf][3][lane];
    acc[0] = wmma_bf16(af, b0, acc[0]);
    acc[1] = wmma_bf16_ra(af, b1, acc[1]);  // A reused across the burst
    acc[2] = wmma_bf16_ra(af, b2, acc[2]);
    acc[3] = wmma_bf16_ra(af, b3, acc[3]);
    if (c + 1 < NC) lstore((c + 1) & 1);
  }

  const long long ob = (long long)b * M * N;
#pragma unroll
  for (int t = 0; t < 4; ++t) {
#pragma unroll
    for (int r = 0; r < 8; ++r) {
      int row = m0 + wave * 16 + r + 8 * half;
      int col = n0 + t * 16 + l16;
      float vv = acc[t][r] + (bias ? bias[row] : 0.0f);
      if (act == 1) vv = 0.5f * vv * (1.0f + erff(vv * 0.70710678118654752f));
      long long idx = ob + (long long)row * N + col;
      if (res) vv += res[idx];
      if (outF) outF[idx] = vv;
      if (outB) outB[idx] = (bf16_t)vv;
    }
  }
}

// ----------------------- fused flash attention (wave32 WMMA, pipelined staging) -------------
// q/k/v channel-major: t[(h*64+d)*ld + token]. Block = 128 queries of one (b,h); wave w owns
// queries [w*16, w*16+16). All key-tile loads are in-bounds (ld >= padded tile width); invalid
// keys are neutralized purely by the softmax mask (S=-inf -> P=0 exactly), so staging has no
// guards. K/V tile kt+64 is global-loaded into registers while tile kt computes.
__global__ void __launch_bounds__(256) k_attention(
    const bf16_t* __restrict__ qp, const bf16_t* __restrict__ kp,
    const bf16_t* __restrict__ vp, bf16_t* __restrict__ op, int Nk, int ldq, int ldk,
    int ldo, long long qbs, long long kbs, long long vbs, long long obs, float scale) {
  __shared__ v16bf Qf[8][2][32];  // [wave][d-chunk][lane] A-frags; reused as O transpose buf
  __shared__ v16bf Kf[4][2][32];  // [key-subtile][d-chunk][lane] B-frags for S
  __shared__ v16bf Vf[4][2][32];  // [d-subtile][key-chunk][lane] B-frags for P@V
  __shared__ v16bf Pf[8][2][32];  // [wave][key-chunk][lane] A-frags of P

  const int tid = threadIdx.x;
  const int lane = tid & 31;
  const int wave = tid >> 5;
  const int half = lane >> 4;
  const int l16 = lane & 15;
  const int b = blockIdx.z;
  const int h = blockIdx.y;
  const int q0 = blockIdx.x * 128;

  const bf16_t* q = qp + (long long)b * qbs + (long long)h * 64 * ldq;
  const bf16_t* k = kp + (long long)b * kbs + (long long)h * 64 * ldk;
  const bf16_t* v = vp + (long long)b * vbs + (long long)h * 64 * ldk;
  bf16_t* o = op + (long long)b * obs + (long long)h * 64 * ldo;

  bf16_t* Qfp = (bf16_t*)Qf;
  bf16_t* Kfp = (bf16_t*)Kf;
  bf16_t* Pfp0 = (bf16_t*)Pf;

  // ---- stage Q once: batched global loads, then permuted LDS stores ----
  {
    uint32_t qreg[16];
#pragma unroll
    for (int j = 0; j < 16; ++j) {
      int i = tid + j * 256;  // pairs along query
      int d = i >> 6, qc2 = (i & 63) * 2;
      qreg[j] = *(const uint32_t*)(q + (long long)d * ldq + q0 + qc2);
    }
#pragma unroll
    for (int j = 0; j < 16; ++j) {
      int i = tid + j * 256;
      int d = i >> 6, qc2 = (i & 63) * 2;
      int kd = d & 31;
      int halfA = (kd >> 3) & 1;
      int e = 8 * (kd >> 4) + (kd & 7);
      int base = ((qc2 >> 4) * 2 + (d >> 5)) * 32;
      U32H u;
      u.u = qreg[j];
      Qfp[(base + (qc2 & 15) + 16 * halfA) * 16 + e] = u.h[0];
      Qfp[(base + ((qc2 + 1) & 15) + 16 * halfA) * 16 + e] = u.h[1];
    }
  }

  uint32_t kreg[8], vreg[8];
  auto gloadKV = [&](int kt) {
#pragma unroll
    for (int j = 0; j < 8; ++j) {
      int i = tid + j * 256;  // pairs along key; always in-bounds memory
      int d = i >> 5, kl2 = (i & 31) * 2;
      int key = kt + kl2;
      kreg[j] = *(const uint32_t*)(k + (long long)d * ldk + key);
      vreg[j] = *(const uint32_t*)(v + (long long)d * ldk + key);
    }
  };
  auto lstoreKV = [&]() {
#pragma unroll
    for (int j = 0; j < 8; ++j) {
      int i = tid + j * 256;
      int d = i >> 5, kl2 = (i & 31) * 2;
      int kd = d & 31;
      int halfB = (kd >> 3) & 1, e = 8 * (kd >> 4) + (kd & 7);  // K: B-frag with k=d, n=key
      int base = ((kl2 >> 4) * 2 + (d >> 5)) * 32;
      U32H u;
      u.u = kreg[j];
      Kfp[(base + (kl2 & 15) + 16 * halfB) * 16 + e] = u.h[0];
      Kfp[(base + ((kl2 + 1) & 15) + 16 * halfB) * 16 + e] = u.h[1];
      // V: B-frag with k=key, n=d -> pairs land in one lane at consecutive e (4B store)
      int halfV = (kl2 >> 3) & 1;
      int eV = 8 * ((kl2 >> 4) & 1) + (kl2 & 7);
      ((uint32_t*)&Vf[d >> 4][kl2 >> 5][(d & 15) + 16 * halfV])[eV >> 1] = vreg[j];
    }
  };

  gloadKV(0);
  __syncthreads();  // Qf visible
  v16bf qfr[2] = {Qf[wave][0][lane], Qf[wave][1][lane]};  // register-resident

  v8f oacc[4] = {};
  float rm[8], rl[8];
#pragma unroll
  for (int r = 0; r < 8; ++r) { rm[r] = -3.0e38f; rl[r] = 0.f; }

  for (int kt = 0; kt < Nk; kt += 64) {
    __syncthreads();  // previous iteration done reading Kf/Vf
    lstoreKV();
    __syncthreads();  // Kf/Vf visible
    if (kt + 64 < Nk) gloadKV(kt + 64);  // next tile in flight during compute

    // S(16q x 64k) = Q @ K^T : d = 2 chunks of 32; A (Q-frag) reused across key subtiles
    v8f sacc[4] = {};
#pragma unroll
    for (int kc = 0; kc < 2; ++kc) {
      sacc[0] = wmma_bf16(qfr[kc], Kf[0][kc][lane], sacc[0]);
      sacc[1] = wmma_bf16_ra(qfr[kc], Kf[1][kc][lane], sacc[1]);
      sacc[2] = wmma_bf16_ra(qfr[kc], Kf[2][kc][lane], sacc[2]);
      sacc[3] = wmma_bf16_ra(qfr[kc], Kf[3][kc][lane], sacc[3]);
    }

    // online softmax (rows live in a half-wave)
#pragma unroll
    for (int r = 0; r < 8; ++r) {
      float mx = -3.0e38f;
#pragma unroll
      for (int t = 0; t < 4; ++t) {
        float sv = sacc[t][r] * scale;
        int key = kt + t * 16 + l16;
        sv = (key < Nk) ? sv : -3.0e38f;  // masks padded/garbage keys exactly (P becomes 0)
        sacc[t][r] = sv;
        mx = fmaxf(mx, sv);
      }
#pragma unroll
      for (int off = 8; off >= 1; off >>= 1) mx = fmaxf(mx, __shfl_xor(mx, off, 32));
      float mnew = fmaxf(rm[r], mx);
      float alpha = __expf(rm[r] - mnew);
      rm[r] = mnew;
      float rs = 0.f;
#pragma unroll
      for (int t = 0; t < 4; ++t) {
        float pv = __expf(sacc[t][r] - mnew);
        sacc[t][r] = pv;
        rs += pv;
      }
#pragma unroll
      for (int off = 8; off >= 1; off >>= 1) rs += __shfl_xor(rs, off, 32);
      rl[r] = rl[r] * alpha + rs;
#pragma unroll
      for (int t2 = 0; t2 < 4; ++t2) oacc[t2][r] *= alpha;
    }

    // P: D-layout -> A-fragment order in per-wave LDS region (in-order DS within a wave)
    bf16_t* Pfp = Pfp0 + wave * (2 * 32 * 16);
#pragma unroll
    for (int t = 0; t < 4; ++t)
#pragma unroll
      for (int r = 0; r < 8; ++r) {
        int row = r + 8 * half;
        int col = t * 16 + l16;
        int kd = col & 31;
        int halfA = (kd >> 3) & 1;
        int e = 8 * (kd >> 4) + (kd & 7);
        Pfp[((col >> 5) * 32 + row + 16 * halfA) * 16 + e] = (bf16_t)sacc[t][r];
      }

    // O += P(16q x 64k) @ V(64k x 64d); P-frag reused across d subtiles
#pragma unroll
    for (int kc = 0; kc < 2; ++kc) {
      v16bf pf = Pf[wave][kc][lane];
      oacc[0] = wmma_bf16(pf, Vf[0][kc][lane], oacc[0]);
      oacc[1] = wmma_bf16_ra(pf, Vf[1][kc][lane], oacc[1]);
      oacc[2] = wmma_bf16_ra(pf, Vf[2][kc][lane], oacc[2]);
      oacc[3] = wmma_bf16_ra(pf, Vf[3][kc][lane], oacc[3]);
    }
  }

  // ---- epilogue: transpose O through (dead) Qf LDS, coalesced 4B global stores ----
  bf16_t* Od = (bf16_t*)Qf;  // 64 d x 128 q
#pragma unroll
  for (int r = 0; r < 8; ++r) {
    float inv = 1.0f / rl[r];
    int rowl = wave * 16 + r + 8 * half;
#pragma unroll
    for (int t2 = 0; t2 < 4; ++t2) {
      int d = t2 * 16 + l16;
      Od[d * 128 + rowl] = (bf16_t)(oacc[t2][r] * inv);
    }
  }
  __syncthreads();
#pragma unroll
  for (int j = 0; j < 16; ++j) {
    int i = tid + j * 256;  // 4096 pairs
    int d = i >> 6, rp = (i & 63) * 2;
    uint32_t vv = *(uint32_t*)(Od + d * 128 + rp);
    *(uint32_t*)(o + (long long)d * ldo + q0 + rp) = vv;
  }
}

// ----------------------- launch -----------------------
extern "C" void kernel_launch(void* const* d_in, const int* in_sizes, int n_in,
                              void* d_out, int out_size, void* d_ws, size_t ws_size,
                              hipStream_t stream) {
  (void)in_sizes; (void)n_in; (void)out_size; (void)ws_size;
  const float* x        = (const float*)d_in[0];
  const float* context  = (const float*)d_in[1];
  const float* sa_nw    = (const float*)d_in[2];
  const float* sa_nb    = (const float*)d_in[3];
  const float* sa_qkv_w = (const float*)d_in[4];
  const float* sa_qkv_b = (const float*)d_in[5];
  const float* sa_pj_w  = (const float*)d_in[6];
  const float* sa_pj_b  = (const float*)d_in[7];
  const float* ca_nw    = (const float*)d_in[8];
  const float* ca_nb    = (const float*)d_in[9];
  const float* ca_q_w   = (const float*)d_in[10];
  const float* ca_q_b   = (const float*)d_in[11];
  const float* ca_kv_w  = (const float*)d_in[12];
  const float* ca_kv_b  = (const float*)d_in[13];
  const float* ca_pj_w  = (const float*)d_in[14];
  const float* ca_pj_b  = (const float*)d_in[15];
  const float* ff_nw    = (const float*)d_in[16];
  const float* ff_nb    = (const float*)d_in[17];
  const float* ff_w1    = (const float*)d_in[18];
  const float* ff_b1    = (const float*)d_in[19];
  const float* ff_w2    = (const float*)d_in[20];
  const float* ff_b2    = (const float*)d_in[21];

  char* p = (char*)d_ws;
  auto take = [&](size_t bytes) {
    char* r = p;
    p += (bytes + 255) & ~(size_t)255;
    return r;
  };
  float*  cur    = (float*) take(4ull * 512 * 1024 * 4);
  bf16_t* hb     = (bf16_t*)take(4ull * 512 * 1024 * 2);
  bf16_t* qkv    = (bf16_t*)take(4ull * 1536 * 1024 * 2);
  bf16_t* attno  = (bf16_t*)take(4ull * 512 * 1024 * 2);
  bf16_t* qbuf   = (bf16_t*)take(4ull * 512 * 1024 * 2);
  bf16_t* ctxT   = (bf16_t*)take(4ull * 768 * 128 * 2);   // padded to 128 tokens
  bf16_t* kvbuf  = (bf16_t*)take(4ull * 1024 * 128 * 2);  // padded to 128 tokens
  bf16_t* ffnbuf = (bf16_t*)take(4ull * 2048 * 1024 * 2);
  bf16_t* w_qkv  = (bf16_t*)take(1536ull * 512 * 2);
  bf16_t* w_sapj = (bf16_t*)take(512ull * 512 * 2);
  bf16_t* w_caq  = (bf16_t*)take(512ull * 512 * 2);
  bf16_t* w_cakv = (bf16_t*)take(1024ull * 768 * 2);
  bf16_t* w_capj = (bf16_t*)take(512ull * 512 * 2);
  bf16_t* w_f1   = (bf16_t*)take(2048ull * 512 * 2);
  bf16_t* w_f2   = (bf16_t*)take(512ull * 2048 * 2);

  auto cg4 = [](int n) { return dim3((unsigned)((n / 4 + 255) / 256)); };
  auto cg  = [](int n) { return dim3((unsigned)((n + 255) / 256)); };

  k_copy_f32v<<<cg4(2097152), 256, 0, stream>>>((const float4*)x, (float4*)cur, 524288);
  k_f32_to_bf16v<<<cg4(786432), 256, 0, stream>>>((const float4*)sa_qkv_w, w_qkv, 196608);
  k_f32_to_bf16v<<<cg4(262144), 256, 0, stream>>>((const float4*)sa_pj_w, w_sapj, 65536);
  k_f32_to_bf16v<<<cg4(262144), 256, 0, stream>>>((const float4*)ca_q_w, w_caq, 65536);
  k_f32_to_bf16v<<<cg4(786432), 256, 0, stream>>>((const float4*)ca_kv_w, w_cakv, 196608);
  k_f32_to_bf16v<<<cg4(262144), 256, 0, stream>>>((const float4*)ca_pj_w, w_capj, 65536);
  k_f32_to_bf16v<<<cg4(1048576), 256, 0, stream>>>((const float4*)ff_w1, w_f1, 262144);
  k_f32_to_bf16v<<<cg4(1048576), 256, 0, stream>>>((const float4*)ff_w2, w_f2, 262144);
  k_ctx_transpose<<<cg(4 * 768 * 128), 256, 0, stream>>>(context, ctxT);

  // ---- self-attention ----
  k_groupnorm<<<32, 256, 0, stream>>>(cur, sa_nw, sa_nb, hb);
  k_gemm<<<dim3(16, 12, 4), 256, 0, stream>>>(w_qkv, hb, sa_qkv_b, nullptr, nullptr, qkv,
                                              1536, 1024, 512, 0);
  k_attention<<<dim3(8, 8, 4), 256, 0, stream>>>(
      qkv, qkv + 512 * 1024, qkv + 1024 * 1024, attno, 1024, 1024, 1024, 1024,
      (long long)1536 * 1024, (long long)1536 * 1024, (long long)1536 * 1024,
      (long long)512 * 1024, 0.125f);
  k_gemm<<<dim3(16, 4, 4), 256, 0, stream>>>(w_sapj, attno, sa_pj_b, cur, cur, nullptr,
                                             512, 1024, 512, 0);

  // ---- cross-attention ----
  k_groupnorm<<<32, 256, 0, stream>>>(cur, ca_nw, ca_nb, hb);
  k_gemm<<<dim3(16, 4, 4), 256, 0, stream>>>(w_caq, hb, ca_q_b, nullptr, nullptr, qbuf,
                                             512, 1024, 512, 0);
  k_gemm<<<dim3(2, 8, 4), 256, 0, stream>>>(w_cakv, ctxT, ca_kv_b, nullptr, nullptr,
                                            kvbuf, 1024, 128, 768, 0);
  k_attention<<<dim3(8, 8, 4), 256, 0, stream>>>(
      qbuf, kvbuf, kvbuf + 512 * 128, attno, 77, 1024, 128, 1024,
      (long long)512 * 1024, (long long)1024 * 128, (long long)1024 * 128,
      (long long)512 * 1024, 0.125f);
  k_gemm<<<dim3(16, 4, 4), 256, 0, stream>>>(w_capj, attno, ca_pj_b, cur, cur, nullptr,
                                             512, 1024, 512, 0);

  // ---- FFN ----
  k_groupnorm<<<32, 256, 0, stream>>>(cur, ff_nw, ff_nb, hb);
  k_gemm<<<dim3(16, 16, 4), 256, 0, stream>>>(w_f1, hb, ff_b1, nullptr, nullptr, ffnbuf,
                                              2048, 1024, 512, 1);
  k_gemm<<<dim3(16, 4, 4), 256, 0, stream>>>(w_f2, ffnbuf, ff_b2, cur, (float*)d_out,
                                             nullptr, 512, 1024, 2048, 0);
}
